// AVRRender_86492051407425
// MI455X (gfx1250) — matrix operand order repaired and checked by reading, not compiled, not run on hardware
//
#include <hip/hip_runtime.h>
#include <hip/hip_bf16.h>
#include <math.h>

typedef __attribute__((ext_vector_type(16))) _Float16 v16h;
typedef __attribute__((ext_vector_type(8)))  float    v8f;

#define SEQ    4096
#define NS     128
#define NF     2049
#define FPAD   2064          // NF padded up to multiple of 16
#define NFT    (FPAD/16)     // 129 freq tiles
#define NRAYS  130
#define BSZ    16
#define MROWS  (BSZ*NS)      // 2048
#define SPEEDC 343.0f
#define FSAMP  48000.0f
#define TWO_PI 6.28318530717958647692f
#define DSTEP  (9.9f/127.0f)
#define S2T    (FSAMP/SPEEDC)     // samples per meter (compile-time constant)
#define T2S    (SPEEDC/FSAMP)

// workspace layout (bytes, all 256B aligned)
#define OFF_W  0u                      // weights  130*2048 f32   = 1,064,960
#define OFF_D  1064960u                // delays   130*2048 f32
#define OFF_Z  2129920u                // z        2048*4096 f16  = 16,777,216
#define OFF_C  18907136u               // DFT cos  2064*4096 f16  = 16,908,288
#define OFF_S  35815424u               // DFT sin  2064*4096 f16
// total ~52.7 MB

union V16H { v16h v; _Float16 h[16]; float4 f4[2]; };

__device__ __forceinline__ float hash01(int i) {
    float x = __sinf((float)i * 12.9898f + 78.233f) * 43758.5453f;
    return x - floorf(x);
}

// deterministic stand-in for _directions() (jax threefry not reproducible on-device)
__device__ __forceinline__ void ray_dir(int r, float& dx, float& dy, float& dz) {
    if (r == 128) { dx = 0.0f; dy = 0.0f; dz = 1.0f;  return; }
    if (r == 129) { dx = 0.0f; dy = 0.0f; dz = -1.0f; return; }
    int ia = r >> 3;          // azimuth index (NELE=8)
    int je = r & 7;           // elevation index
    float a  = ((float)ia + hash01(ia)) * (TWO_PI / 16.0f);
    float el = (float)(je + 1) * (1.0f / 9.0f);
    float e  = acosf(2.0f * el - 1.0f);
    float se = __sinf(e);
    dx = __cosf(a) * se;
    dy = __sinf(a) * se;
    dz = __cosf(e);
}

// branch-free fast tanh: tanh(x) = 1 - 2/(exp(2x)+1); rcp via HW v_rcp_f32
__device__ __forceinline__ float tanh_fast(float x) {
    float cx = fminf(fmaxf(x, -9.0f), 9.0f);
    float e2 = __expf(2.0f * cx);
    return 1.0f - 2.0f * __builtin_amdgcn_rcpf(e2 + 1.0f);
}

__device__ __forceinline__ float softplus_f(float x) {
    return (x > 20.0f) ? x : __logf(1.0f + __expf(x));
}

__device__ __forceinline__ float pl_of(int i) {
    float ii = (i < 13) ? 14.0f : (float)i;     // pl[:13] = pl[14]
    return __builtin_amdgcn_rcpf(ii * T2S + 0.001f);
}

// ---------------------------------------------------------------------------
// K_zero: clear the complex64 output (bs x F, interleaved re/im)
// ---------------------------------------------------------------------------
__global__ void avr_zero_kernel(float* __restrict__ out, int n) {
    int i = blockIdx.x * blockDim.x + threadIdx.x;
    if (i < n) out[i] = 0.0f;
}

// ---------------------------------------------------------------------------
// K_gen: DFT twiddle tables, stored transposed [f][t] so WMMA-B loads are
// contiguous 16B chunks.  Exact integer range reduction: (f*t) mod 4096.
// ---------------------------------------------------------------------------
__global__ void avr_tables_kernel(_Float16* __restrict__ Ct, _Float16* __restrict__ St) {
    size_t idx = (size_t)blockIdx.x * blockDim.x + threadIdx.x;
    if (idx >= (size_t)FPAD * SEQ) return;
    int f  = (int)(idx >> 12);
    int t  = (int)(idx & 4095);
    int ft = (f * t) & 4095;
    float ang = -(TWO_PI / (float)SEQ) * (float)ft;
    Ct[idx] = (_Float16)__cosf(ang);
    St[idx] = (_Float16)__sinf(ang);
}

// ---------------------------------------------------------------------------
// K0: alpha-compositing weights + tx->point delays; one thread per (ray,b)
// scans the 128 samples sequentially (cumprod).
// ---------------------------------------------------------------------------
__global__ void __launch_bounds__(256)
avr_weights_kernel(const float* __restrict__ rays_o,
                   const float* __restrict__ position_tx,
                   const float* __restrict__ direction_tx,
                   const float* __restrict__ w_attn,
                   const float* __restrict__ b_attn,
                   float* __restrict__ w_ws,
                   float* __restrict__ delay_ws) {
    int tid = blockIdx.x * blockDim.x + threadIdx.x;
    if (tid >= NRAYS * BSZ) return;
    int r = tid >> 4;
    int b = tid & 15;

    float dx, dy, dz; ray_dir(r, dx, dy, dz);
    float ox = rays_o[b*3+0], oy = rays_o[b*3+1], oz = rays_o[b*3+2];
    float tx = position_tx[b*3+0], ty = position_tx[b*3+1], tz = position_tx[b*3+2];
    float ux = direction_tx[b*3+0], uy = direction_tx[b*3+1], uz = direction_tx[b*3+2];

    float wa[12];
    #pragma unroll
    for (int k = 0; k < 12; ++k) wa[k] = w_attn[k];
    // _norm_pts(p) == p/5 exactly (XMIN=-5, XMAX=5)
    float base = (-dx)*wa[3] + (-dy)*wa[4] + (-dz)*wa[5]
               + tx*0.2f*wa[6] + ty*0.2f*wa[7] + tz*0.2f*wa[8]
               + ux*wa[9] + uy*wa[10] + uz*wa[11] + b_attn[0];

    float att = 1.0f;
    for (int s = 0; s < NS; ++s) {
        float ds = 0.1f + (float)s * DSTEP;
        float px = ox + dx*ds, py = oy + dy*ds, pz = oz + dz*ds;
        float dot = base + px*0.2f*wa[0] + py*0.2f*wa[1] + pz*0.2f*wa[2];
        float attn = softplus_f(dot);
        float dist = (s == NS-1) ? 1e10f : DSTEP;
        float alpha = 1.0f - __expf(-attn * dist);
        float w = att * alpha;
        att *= (1.0f - alpha + 1e-6f);
        // _denorm_pts(tx_n - pts_n) == position_tx - ray_pts exactly
        float rx = tx - px, ry = ty - py, rz = tz - pz;
        float t2p = sqrtf(rx*rx + ry*ry + rz*rz) * S2T;
        float delay = fminf(fmaxf(roundf(t2p), 0.0f), (float)(SEQ - 1));
        size_t o = (size_t)r * MROWS + (size_t)b * NS + s;
        w_ws[o]     = w;
        delay_ws[o] = delay;
    }
}

// ---------------------------------------------------------------------------
// K1: per-ray MLP via WMMA, tanh + masks + weight, accumulated over 130 rays.
// Block = 256 thr (8 waves); block tile = 16 rows (b,s) x 256 cols (t);
// each wave: 16x32 cols -> 2 WMMAs/ray.  B (=w_sig, K=12 pad 32) is
// ray-invariant and lives in registers for the whole ray loop.
// ---------------------------------------------------------------------------
__global__ void __launch_bounds__(256)
avr_accum_kernel(const float* __restrict__ rays_o,
                 const float* __restrict__ position_tx,
                 const float* __restrict__ direction_tx,
                 const float* __restrict__ w_sig,
                 const float* __restrict__ w_ws,
                 const float* __restrict__ delay_ws,
                 _Float16* __restrict__ z) {
    const int lane = threadIdx.x & 31;
    const int wid  = threadIdx.x >> 5;
    const int rtb  = blockIdx.x * 16;          // row-tile base in m = b*128+s
    const int b    = rtb >> 7;                 // constant per block (tiles never cross b)
    const int n0   = blockIdx.y * 256 + wid * 32;
    const int n1   = n0 + 16;
    const int col  = lane & 15;

    // ---- B tiles: lanes 0-15 hold K=0..15 (K=12 valid), lanes 16-31 are K=16..31 = 0
    V16H b0, b1;
    #pragma unroll
    for (int k = 0; k < 16; ++k) { b0.h[k] = (_Float16)0.0f; b1.h[k] = (_Float16)0.0f; }
    if (lane < 16) {
        #pragma unroll
        for (int k = 0; k < 12; ++k) {
            b0.h[k] = (_Float16)w_sig[k * SEQ + n0 + col];
            b1.h[k] = (_Float16)w_sig[k * SEQ + n1 + col];
        }
    }

    // batch-constant features
    const float ox = rays_o[b*3+0], oy = rays_o[b*3+1], oz = rays_o[b*3+2];
    const float tx = position_tx[b*3+0], ty = position_tx[b*3+1], tz = position_tx[b*3+2];
    const float ux = direction_tx[b*3+0], uy = direction_tx[b*3+1], uz = direction_tx[b*3+2];

    // A-row sample depth for this lane (row M = lane&15)
    const float dsA = 0.1f + (float)((rtb & 127) + col) * DSTEP;

    // accumulator rows for this lane: off..off+7 (C layout: VGPR j -> M=j or j+8)
    const int off = (lane & 16) ? 8 : 0;
    float shiftJ[8];
    #pragma unroll
    for (int j = 0; j < 8; ++j) {
        int sj = (rtb & 127) + off + j;
        shiftJ[j] = roundf((0.1f + (float)sj * DSTEP) * S2T);
    }

    v8f acc0 = {}; v8f acc1 = {};

    for (int r = 0; r < NRAYS; ++r) {
        float dx, dy, dz; ray_dir(r, dx, dy, dz);

        // ---- A tile (16x32 f16): lanes 0-15 -> K0..11, lanes 16-31 -> 0
        V16H a;
        #pragma unroll
        for (int k = 0; k < 16; ++k) a.h[k] = (_Float16)0.0f;
        if (lane < 16) {
            float px = (ox + dx * dsA) * 0.2f;
            float py = (oy + dy * dsA) * 0.2f;
            float pz = (oz + dz * dsA) * 0.2f;
            a.h[0] = (_Float16)px;        a.h[1] = (_Float16)py;
            a.h[2] = (_Float16)pz;        a.h[3] = (_Float16)(-dx);
            a.h[4] = (_Float16)(-dy);     a.h[5] = (_Float16)(-dz);
            a.h[6] = (_Float16)(tx*0.2f); a.h[7] = (_Float16)(ty*0.2f);
            a.h[8] = (_Float16)(tz*0.2f);
            a.h[9] = (_Float16)ux; a.h[10] = (_Float16)uy; a.h[11] = (_Float16)uz;
        }

        // ---- per-row weight + delay for the 8 rows this lane accumulates
        const float* wp = w_ws     + (size_t)r * MROWS + rtb + off;
        const float* dp = delay_ws + (size_t)r * MROWS + rtb + off;
        float wv[8], dv[8];
        *(float4*)&wv[0] = *(const float4*)&wp[0];
        *(float4*)&wv[4] = *(const float4*)&wp[4];
        *(float4*)&dv[0] = *(const float4*)&dp[0];
        *(float4*)&dv[4] = *(const float4*)&dp[4];

        v8f c0 = {}; v8f c1 = {};
        c0 = __builtin_amdgcn_wmma_f32_16x16x32_f16(false, a.v, false, b0.v, (short)0, c0, false, false);
        c1 = __builtin_amdgcn_wmma_f32_16x16x32_f16(false, a.v, false, b1.v, (short)0, c1, false, false);

        float t0 = (float)(n0 + col);
        float t1 = (float)(n1 + col);
        #pragma unroll
        for (int j = 0; j < 8; ++j) {
            float tail = 4095.0f - shiftJ[j];             // mask_tail: t < 4095 - shift
            float v0 = tanh_fast(c0[j]);
            float v1 = tanh_fast(c1[j]);
            // select weight (0 when masked) -> single fma per element, no branches
            float w0 = ((t0 < tail) && (t0 >= dv[j])) ? wv[j] : 0.0f;
            float w1 = ((t1 < tail) && (t1 >= dv[j])) ? wv[j] : 0.0f;
            acc0[j] = fmaf(w0, v0, acc0[j]);
            acc1[j] = fmaf(w1, v1, acc1[j]);
        }
    }

    // epilogue: path loss, store z as f16
    #pragma unroll
    for (int j = 0; j < 8; ++j) {
        int mj = rtb + off + j;
        int t0 = n0 + col, t1 = n1 + col;
        int sh = (int)shiftJ[j];
        z[(size_t)mj * SEQ + t0] = (_Float16)(acc0[j] * pl_of(sh + t0));
        z[(size_t)mj * SEQ + t1] = (_Float16)(acc1[j] * pl_of(sh + t1));
    }
}

// ---------------------------------------------------------------------------
// K2: rfft-as-GEMM.  Z[m,f] = sum_t z[m,t]*exp(-i*2pi*f*t/4096) via two f16
// WMMA chains (cos/sin), K=4096 in steps of 32; epilogue applies the
// fractional-delay phase and reduces over samples s with HW f32 atomics.
// ---------------------------------------------------------------------------
__global__ void __launch_bounds__(256)
avr_dft_kernel(const _Float16* __restrict__ z,
               const _Float16* __restrict__ Ct,
               const _Float16* __restrict__ St,
               float* __restrict__ out) {
    const int lane  = threadIdx.x & 31;
    const int wid   = threadIdx.x >> 5;
    const int rtb   = blockIdx.x * 16;
    const int ftile = blockIdx.y * 8 + wid;
    if (ftile >= NFT) return;                  // wave-uniform: EXEC stays all-ones
    const int fbase = ftile * 16;
    const int col   = lane & 15;
    const int f     = fbase + col;

    const _Float16* zrow = z  + (size_t)(rtb + col) * SEQ;
    const _Float16* crow = Ct + (size_t)f * SEQ;
    const _Float16* srow = St + (size_t)f * SEQ;
    const int aoff = (lane & 16) ? 16 : 0;     // A: K k..k+15 / k+16..k+31
    const int boff = (lane & 16) ? 16 : 0;     // B: K k..k+15 / k+16..k+31

    v8f cre = {}; v8f cim = {};
    #pragma unroll 2
    for (int k = 0; k < SEQ; k += 32) {
        V16H a, bc, bs;
        *(float4*)&a.h[0]  = *(const float4*)&zrow[k + aoff];
        *(float4*)&a.h[8]  = *(const float4*)&zrow[k + aoff + 8];
        *(float4*)&bc.h[0] = *(const float4*)&crow[k + boff];
        *(float4*)&bc.h[8] = *(const float4*)&crow[k + boff + 8];
        *(float4*)&bs.h[0] = *(const float4*)&srow[k + boff];
        *(float4*)&bs.h[8] = *(const float4*)&srow[k + boff + 8];
        __builtin_prefetch(&zrow[k + 256], 0, 1);
        cre = __builtin_amdgcn_wmma_f32_16x16x32_f16(false, a.v, false, bc.v, (short)0, cre, false, false);
        cim = __builtin_amdgcn_wmma_f32_16x16x32_f16(false, a.v, false, bs.v, (short)0, cim, false, false);
    }

    if (f < NF) {
        #pragma unroll
        for (int j = 0; j < 8; ++j) {
            int mj = rtb + ((lane & 16) ? 8 : 0) + j;
            int s  = mj & 127;
            int bb = mj >> 7;
            float p2r = (0.1f + (float)s * DSTEP) * S2T;
            float ang = -(TWO_PI / (float)SEQ) * (float)f * p2r;
            ang -= TWO_PI * floorf(ang * (1.0f / TWO_PI));   // range reduce
            float cc = __cosf(ang), ss = __sinf(ang);
            float re = cre[j], im = cim[j];
            size_t o = ((size_t)bb * NF + f) * 2;
            unsafeAtomicAdd(&out[o + 0], re * cc - im * ss);
            unsafeAtomicAdd(&out[o + 1], re * ss + im * cc);
        }
    }
}

// ---------------------------------------------------------------------------
extern "C" void kernel_launch(void* const* d_in, const int* in_sizes, int n_in,
                              void* d_out, int out_size, void* d_ws, size_t ws_size,
                              hipStream_t stream) {
    (void)in_sizes; (void)n_in; (void)out_size; (void)ws_size;
    const float* rays_o = (const float*)d_in[0];
    const float* ptx    = (const float*)d_in[1];
    const float* dtx    = (const float*)d_in[2];
    const float* w_attn = (const float*)d_in[3];
    const float* b_attn = (const float*)d_in[4];
    const float* w_sig  = (const float*)d_in[5];
    float* out = (float*)d_out;

    char* ws = (char*)d_ws;
    float*    w_ws     = (float*)(ws + OFF_W);
    float*    delay_ws = (float*)(ws + OFF_D);
    _Float16* z        = (_Float16*)(ws + OFF_Z);
    _Float16* Ct       = (_Float16*)(ws + OFF_C);
    _Float16* St       = (_Float16*)(ws + OFF_S);

    int outn = BSZ * NF * 2;
    avr_zero_kernel<<<(outn + 255) / 256, 256, 0, stream>>>(out, outn);

    int tabN = (int)(((size_t)FPAD * SEQ + 255) / 256);
    avr_tables_kernel<<<tabN, 256, 0, stream>>>(Ct, St);

    avr_weights_kernel<<<(NRAYS * BSZ + 255) / 256, 256, 0, stream>>>(
        rays_o, ptx, dtx, w_attn, b_attn, w_ws, delay_ws);

    avr_accum_kernel<<<dim3(MROWS / 16, SEQ / 256), 256, 0, stream>>>(
        rays_o, ptx, dtx, w_sig, w_ws, delay_ws, z);

    avr_dft_kernel<<<dim3(MROWS / 16, (NFT + 7) / 8), 256, 0, stream>>>(
        z, Ct, St, out);
}
